// Hawk_89438398972453
// MI455X (gfx1250) — matrix operand order repaired and checked by reading, not compiled, not run on hardware
//
#include <hip/hip_runtime.h>
#include <hip/hip_bf16.h>
#include <math.h>

// ---------------- problem constants ----------------
#define DIMC   1024
#define HIDC   1536
#define KCONV  4
#define NBATCH 4
#define TSEQ   4096
#define NTROWS (NBATCH * TSEQ)   // 16384 rows (n,t) flattened
#define NCH    64                // scan chunks per sequence
#define CLEN   (TSEQ / NCH)      // 64 steps per chunk

// ---------------- types ----------------
typedef __bf16 bf16_t;
typedef __attribute__((ext_vector_type(16))) __bf16 v16bf;
typedef __attribute__((ext_vector_type(8)))  __bf16 bf16x8;
typedef __attribute__((ext_vector_type(4)))  __bf16 bf16x4;
typedef __attribute__((ext_vector_type(8)))  float  v8f;

// ---------------- GEMM tiling ----------------
#define BM  128
#define BN  128
#define BK  32
#define BKP 56   // padded LDS row stride: 112B rows -> 16B aligned, bank-friendly

__device__ __forceinline__ float sigmoidf_(float x) { return 1.0f / (1.0f + expf(-x)); }

// ============================================================
// Tiled bf16 WMMA GEMM: C[M,N] = A[M,K] * W[N,K]^T  (+ fused epilogue)
// Double-buffered LDS (ping-pong), one barrier per K-step.
// MODE 0: A = x (f32).   cols < HID -> gelu(gate) as bf16 (ob); else u_pre f32 (o0)
// MODE 1: A = u (f32).   +bg; cols < HID -> alpha f32 (o0); else p = sig(inp)*u (o1)
// MODE 2: A = gh (bf16). plain f32 store to o0 (final output)
// 256 threads = 8 waves (2x4); each wave owns a 64x32 tile = 4x2 wmma frags.
// ============================================================
template <int MODE, typename AT>
__global__ __launch_bounds__(256)
void wmma_gemm(const AT* __restrict__ A, int lda,
               const float* __restrict__ W, int ldw, int Kd,
               float* __restrict__ o0, float* __restrict__ o1,
               bf16_t* __restrict__ ob,
               const float* __restrict__ xu,   // u buffer (MODE 1)
               const float* __restrict__ bg,   // gate bias (MODE 1)
               const float* __restrict__ fb)   // forget_base (MODE 1)
{
    __shared__ __align__(16) bf16_t sA[2][BM * BKP];
    __shared__ __align__(16) bf16_t sB[2][BN * BKP];

    const int tid  = threadIdx.x;
    const int wave = tid >> 5;
    const int lane = tid & 31;
    const int half = lane >> 4;   // 0/1 (lane half per WMMA layout)
    const int r    = lane & 15;
    const int wm   = wave >> 2;   // 0..1  (M wave index)
    const int wn   = wave & 3;    // 0..3  (N wave index)
    const int bm0  = blockIdx.x * BM;
    const int bn0  = blockIdx.y * BN;

    v8f acc[4][2];
#pragma unroll
    for (int i = 0; i < 4; ++i)
#pragma unroll
        for (int j = 0; j < 2; ++j)
#pragma unroll
            for (int e = 0; e < 8; ++e) acc[i][j][e] = 0.0f;

    // staged (in-register) raw tiles for the next LDS buffer
    float4 rfa[4];   // A tile, f32 source path
    bf16x8 rba[2];   // A tile, bf16 source path
    float4 rfb[4];   // W tile (always f32)

    auto load_tile = [&](int k0) {
        if constexpr (sizeof(AT) == 4) {
            const int cg = tid & 7, row0 = tid >> 3;
#pragma unroll
            for (int it = 0; it < 4; ++it)
                rfa[it] = *reinterpret_cast<const float4*>(
                    A + (size_t)(bm0 + row0 + it * 32) * lda + k0 + cg * 4);
        } else {
            const int cg = tid & 3, row0 = tid >> 2;
#pragma unroll
            for (int it = 0; it < 2; ++it)
                rba[it] = *reinterpret_cast<const bf16x8*>(
                    A + (size_t)(bm0 + row0 + it * 64) * lda + k0 + cg * 8);
        }
        const int cg = tid & 7, row0 = tid >> 3;
#pragma unroll
        for (int it = 0; it < 4; ++it)
            rfb[it] = *reinterpret_cast<const float4*>(
                W + (size_t)(bn0 + row0 + it * 32) * ldw + k0 + cg * 4);
    };

    auto store_tile = [&](int buf) {
        bf16_t* a_ = &sA[buf][0];
        bf16_t* b_ = &sB[buf][0];
        if constexpr (sizeof(AT) == 4) {
            const int cg = tid & 7, row0 = tid >> 3;
#pragma unroll
            for (int it = 0; it < 4; ++it) {
                bf16x4 b;
                b[0] = (bf16_t)rfa[it].x; b[1] = (bf16_t)rfa[it].y;
                b[2] = (bf16_t)rfa[it].z; b[3] = (bf16_t)rfa[it].w;
                *reinterpret_cast<bf16x4*>(&a_[(row0 + it * 32) * BKP + cg * 4]) = b;
            }
        } else {
            const int cg = tid & 3, row0 = tid >> 2;
#pragma unroll
            for (int it = 0; it < 2; ++it)
                *reinterpret_cast<bf16x8*>(&a_[(row0 + it * 64) * BKP + cg * 8]) = rba[it];
        }
        const int cg = tid & 7, row0 = tid >> 3;
#pragma unroll
        for (int it = 0; it < 4; ++it) {
            bf16x4 b;
            b[0] = (bf16_t)rfb[it].x; b[1] = (bf16_t)rfb[it].y;
            b[2] = (bf16_t)rfb[it].z; b[3] = (bf16_t)rfb[it].w;
            *reinterpret_cast<bf16x4*>(&b_[(row0 + it * 32) * BKP + cg * 4]) = b;
        }
    };

    const int nk = Kd / BK;
    load_tile(0);
    store_tile(0);
    __syncthreads();

    for (int ki = 0; ki < nk; ++ki) {
        const int cur = ki & 1;
        if (ki + 1 < nk) load_tile((ki + 1) * BK);
        if (ki + 2 < nk) {  // far prefetch hint -> global_prefetch_b8
            __builtin_prefetch(A + (size_t)(bm0 + (tid >> 1)) * lda +
                                   (ki + 2) * BK + (tid & 1) * 16, 0, 0);
            __builtin_prefetch(W + (size_t)(bn0 + (tid >> 1)) * ldw +
                                   (ki + 2) * BK + (tid & 1) * 16, 0, 0);
        }

        // ---- build fragments per ISA VGPR layouts & 8 WMMAs ----
        {
            const bf16_t* a_ = &sA[cur][0];
            const bf16_t* b_ = &sB[cur][0];
            v16bf afrag[4], bfrag[2];
#pragma unroll
            for (int i = 0; i < 4; ++i) {  // A 16x32: lane row = m, k chunks {8h, 16+8h}
                const int row = wm * 64 + i * 16 + r;
                union { v16bf v; bf16x8 h2[2]; } u_;
                u_.h2[0] = *reinterpret_cast<const bf16x8*>(&a_[row * BKP + half * 8]);
                u_.h2[1] = *reinterpret_cast<const bf16x8*>(&a_[row * BKP + 16 + half * 8]);
                afrag[i] = u_.v;
            }
#pragma unroll
            for (int j = 0; j < 2; ++j) {  // B 32x16: lane col = n, k chunks {8h, 16+8h}
                const int row = wn * 32 + j * 16 + r;
                union { v16bf v; bf16x8 h2[2]; } u_;
                u_.h2[0] = *reinterpret_cast<const bf16x8*>(&b_[row * BKP + half * 8]);
                u_.h2[1] = *reinterpret_cast<const bf16x8*>(&b_[row * BKP + 16 + half * 8]);
                bfrag[j] = u_.v;
            }
#pragma unroll
            for (int i = 0; i < 4; ++i)
#pragma unroll
                for (int j = 0; j < 2; ++j)
                    acc[i][j] = __builtin_amdgcn_wmma_f32_16x16x32_bf16(
                        false, afrag[i], false, bfrag[j],
                        (short)0, acc[i][j], false, false);
        }

        if (ki + 1 < nk) {
            store_tile(cur ^ 1);   // safe: other buffer last read before prev barrier
            __syncthreads();
        }
    }

    // ---- fused epilogues (C layout: m = 8*half + vgpr, n = r); block-uniform split ----
    const int rb0 = bm0 + wm * 64 + half * 8;
    const int cb0 = bn0 + wn * 32 + r;
    if constexpr (MODE == 0) {
        if (bn0 < HIDC) {          // gate half -> exact-erf gelu, bf16
#pragma unroll
            for (int i = 0; i < 4; ++i)
#pragma unroll
                for (int j = 0; j < 2; ++j)
#pragma unroll
                    for (int vi = 0; vi < 8; ++vi) {
                        const float v = acc[i][j][vi];
                        const float g = 0.5f * v * (1.0f + erff(v * 0.70710678118f));
                        ob[(size_t)(rb0 + i * 16 + vi) * HIDC + (cb0 + j * 16)] = (bf16_t)g;
                    }
        } else {                   // u half -> f32 pre-conv buffer
#pragma unroll
            for (int i = 0; i < 4; ++i)
#pragma unroll
                for (int j = 0; j < 2; ++j)
#pragma unroll
                    for (int vi = 0; vi < 8; ++vi)
                        o0[(size_t)(rb0 + i * 16 + vi) * HIDC + (cb0 + j * 16 - HIDC)] =
                            acc[i][j][vi];
        }
    } else if constexpr (MODE == 1) {
        if (bn0 < HIDC) {          // forget half -> alpha
            float m8sp[2];
#pragma unroll
            for (int j = 0; j < 2; ++j)
                m8sp[j] = -8.0f * log1pf(expf(fb[cb0 + j * 16]));
#pragma unroll
            for (int i = 0; i < 4; ++i)
#pragma unroll
                for (int j = 0; j < 2; ++j) {
                    const float bgi = bg[cb0 + j * 16];
#pragma unroll
                    for (int vi = 0; vi < 8; ++vi) {
                        const float v = acc[i][j][vi] + bgi;
                        o0[(size_t)(rb0 + i * 16 + vi) * HIDC + (cb0 + j * 16)] =
                            expf(m8sp[j] * sigmoidf_(v));
                    }
                }
        } else {                   // input half -> p = sigmoid(inp)*u
#pragma unroll
            for (int i = 0; i < 4; ++i)
#pragma unroll
                for (int j = 0; j < 2; ++j) {
                    const float bgi = bg[cb0 + j * 16];
#pragma unroll
                    for (int vi = 0; vi < 8; ++vi) {
                        const float v = acc[i][j][vi] + bgi;
                        const size_t ix =
                            (size_t)(rb0 + i * 16 + vi) * HIDC + (cb0 + j * 16 - HIDC);
                        o1[ix] = sigmoidf_(v) * xu[ix];
                    }
                }
        }
    } else {
#pragma unroll
        for (int i = 0; i < 4; ++i)
#pragma unroll
            for (int j = 0; j < 2; ++j)
#pragma unroll
                for (int vi = 0; vi < 8; ++vi)
                    o0[(size_t)(rb0 + i * 16 + vi) * DIMC + (cb0 + j * 16)] =
                        acc[i][j][vi];
    }
}

// ============================================================
// Causal depthwise conv1d, K=4, layout [n,t,h] (coalesced over h)
// ============================================================
__global__ __launch_bounds__(256)
void dwconv_kernel(const float* __restrict__ u_pre,
                   const float* __restrict__ w,   // [HID, 1, K]
                   const float* __restrict__ b,   // [HID]
                   float* __restrict__ u)
{
    const size_t idx = (size_t)blockIdx.x * 256 + threadIdx.x;  // over NT*HID
    const int h = (int)(idx % HIDC);
    const int t = (int)((idx / HIDC) % TSEQ);
    float acc = b[h];
#pragma unroll
    for (int j = 0; j < KCONV; ++j) {
        const int ts = t - (KCONV - 1) + j;
        if (ts >= 0)
            acc += w[h * KCONV + j] * u_pre[idx + (size_t)(j - (KCONV - 1)) * HIDC];
    }
    u[idx] = acc;
}

// ============================================================
// 3-pass chunked linear-recurrence scan: h_t = a_t h_{t-1} + beta_t p_t
// ============================================================
__global__ __launch_bounds__(256)
void scan_chunks(const float* __restrict__ alpha, const float* __restrict__ p,
                 float* __restrict__ cA, float* __restrict__ cH)
{
    const int h = blockIdx.x * 256 + threadIdx.x;  // gridDim.x = HID/256
    const int c = blockIdx.y, n = blockIdx.z;
    const size_t base = ((size_t)n * TSEQ + (size_t)c * CLEN) * HIDC + h;
    float Ap = 1.0f, hv = 0.0f;
    for (int t = 0; t < CLEN; ++t) {
        const float a  = alpha[base + (size_t)t * HIDC];
        const float pp = p[base + (size_t)t * HIDC];
        const float beta = sqrtf(1.0f - a * a + 1e-6f);
        hv = a * hv + beta * pp;
        Ap *= a;
    }
    const size_t ci = ((size_t)n * NCH + c) * HIDC + h;
    cA[ci] = Ap;
    cH[ci] = hv;
}

__global__ __launch_bounds__(256)
void scan_combine(const float* __restrict__ cA, const float* __restrict__ cH,
                  float* __restrict__ H0)
{
    const int idx = blockIdx.x * 256 + threadIdx.x;  // N*HID threads
    const int n = idx / HIDC, h = idx % HIDC;
    float s = 0.0f;
    for (int c = 0; c < NCH; ++c) {
        const size_t ci = ((size_t)n * NCH + c) * HIDC + h;
        H0[ci] = s;                  // h entering chunk c
        s = cA[ci] * s + cH[ci];
    }
}

__global__ __launch_bounds__(256)
void scan_final(const float* __restrict__ alpha, const float* __restrict__ p,
                const float* __restrict__ H0, const bf16_t* __restrict__ gate_g,
                bf16_t* __restrict__ gh)
{
    const int h = blockIdx.x * 256 + threadIdx.x;
    const int c = blockIdx.y, n = blockIdx.z;
    const size_t base = ((size_t)n * TSEQ + (size_t)c * CLEN) * HIDC + h;
    float hv = H0[((size_t)n * NCH + c) * HIDC + h];
    for (int t = 0; t < CLEN; ++t) {
        const size_t ix = base + (size_t)t * HIDC;
        const float a  = alpha[ix];
        const float pp = p[ix];
        const float beta = sqrtf(1.0f - a * a + 1e-6f);
        hv = a * hv + beta * pp;
        gh[ix] = (bf16_t)((float)gate_g[ix] * hv);   // fused gelu(gate)*h -> bf16
    }
}

// ============================================================
extern "C" void kernel_launch(void* const* d_in, const int* in_sizes, int n_in,
                              void* d_out, int out_size, void* d_ws, size_t ws_size,
                              hipStream_t stream)
{
    (void)in_sizes; (void)n_in; (void)out_size; (void)ws_size;
    const float* x  = (const float*)d_in[0];
    const float* Wi = (const float*)d_in[1];
    const float* cw = (const float*)d_in[2];
    const float* cb = (const float*)d_in[3];
    const float* Wg = (const float*)d_in[4];
    const float* bg = (const float*)d_in[5];
    const float* fb = (const float*)d_in[6];
    const float* Wo = (const float*)d_in[7];
    float* out = (float*)d_out;

    char* ws = (char*)d_ws;
    size_t off = 0;
    auto take = [&](size_t bytes) -> char* {
        char* ptr = ws + off;
        off += (bytes + 255) & ~(size_t)255;
        return ptr;
    };
    bf16_t* gate_g = (bf16_t*)take((size_t)NTROWS * HIDC * 2);
    float*  u_pre  = (float*) take((size_t)NTROWS * HIDC * 4);
    float*  u      = (float*) take((size_t)NTROWS * HIDC * 4);
    float*  alpha  = (float*) take((size_t)NTROWS * HIDC * 4);
    float*  pbuf   = (float*) take((size_t)NTROWS * HIDC * 4);
    float*  cA     = (float*) take((size_t)NBATCH * NCH * HIDC * 4);
    float*  cH     = (float*) take((size_t)NBATCH * NCH * HIDC * 4);
    float*  H0     = (float*) take((size_t)NBATCH * NCH * HIDC * 4);
    bf16_t* gh     = (bf16_t*)take((size_t)NTROWS * HIDC * 2);

    const dim3 blk(256);

    // GEMM1: gi = x * Wi^T -> gelu(gate) bf16 + u_pre f32
    wmma_gemm<0, float><<<dim3(NTROWS / BM, (2 * HIDC) / BN), blk, 0, stream>>>(
        x, DIMC, Wi, DIMC, DIMC, u_pre, nullptr, gate_g, nullptr, nullptr, nullptr);

    // depthwise causal conv
    dwconv_kernel<<<(unsigned)(((size_t)NTROWS * HIDC) / 256), blk, 0, stream>>>(
        u_pre, cw, cb, u);

    // GEMM2: fg = u * Wg^T + bg -> alpha, p (fused gate math)
    wmma_gemm<1, float><<<dim3(NTROWS / BM, (2 * HIDC) / BN), blk, 0, stream>>>(
        u, HIDC, Wg, HIDC, HIDC, alpha, pbuf, nullptr, u, bg, fb);

    // chunked linear recurrence scan, fused gelu(gate)*h
    scan_chunks<<<dim3(HIDC / 256, NCH, NBATCH), blk, 0, stream>>>(alpha, pbuf, cA, cH);
    scan_combine<<<(NBATCH * HIDC) / 256, blk, 0, stream>>>(cA, cH, H0);
    scan_final<<<dim3(HIDC / 256, NCH, NBATCH), blk, 0, stream>>>(
        alpha, pbuf, H0, gate_g, gh);

    // GEMM3: out = gh * Wo^T
    wmma_gemm<2, bf16_t><<<dim3(NTROWS / BM, DIMC / BN), blk, 0, stream>>>(
        gh, HIDC, Wo, HIDC, HIDC, out, nullptr, nullptr, nullptr, nullptr, nullptr);
}
